// GuidedConvolution_63453846831309
// MI455X (gfx1250) — matrix operand order repaired and checked by reading, not compiled, not run on hardware
//
#include <hip/hip_runtime.h>
#include <hip/hip_bf16.h>

typedef __attribute__((ext_vector_type(16))) _Float16 v16h;
typedef __attribute__((ext_vector_type(8)))  _Float16 h8;
typedef __attribute__((ext_vector_type(8)))  float    v8f;

#define NB   8
#define NC   64          // CIN == COUT == GC
#define HH   320
#define WW   320
#define HK   322         // H + 2P
#define WK   322
#define NPIX (HH*WW)     // 102400
#define KCONV 576        // 9 taps * 64 ic (tap-major: k = (ky*3+kx)*64 + ic)
#define NTOT (NB*NC*NPIX) // 52,428,800

__device__ __forceinline__ v16h cat8(h8 lo, h8 hi) {
    return __builtin_shufflevector(lo, hi, 0,1,2,3,4,5,6,7,8,9,10,11,12,13,14,15);
}

// ---------------- kernel 0: zero BN stats ----------------
__global__ void zero_stats_kernel(float* p) {
    if (threadIdx.x < 128) p[threadIdx.x] = 0.0f;
}

// ---------------- kernel 1: mean over H,W per (b,g) ----------------
__global__ __launch_bounds__(256) void meang_kernel(const float* __restrict__ g,
                                                    float* __restrict__ meanG) {
    __shared__ float red[256];
    const int bc = blockIdx.x;                       // b*64+g
    const float* src = g + (size_t)bc * NPIX;
    float s = 0.f;
    for (int i = threadIdx.x; i < NPIX; i += 256) s += src[i];
    red[threadIdx.x] = s;
    __syncthreads();
    for (int st = 128; st > 0; st >>= 1) {
        if (threadIdx.x < st) red[threadIdx.x] += red[threadIdx.x + st];
        __syncthreads();
    }
    if (threadIdx.x == 0) meanG[bc] = red[0] * (1.0f / (float)NPIX);
}

// ---------------- kernel 2: kcc[b][o][c] = meanG[b]·w_cc[o*64+c][:] + b_cc ----------------
__global__ void kcc_kernel(const float* __restrict__ meanG,
                           const float* __restrict__ w_cc,
                           const float* __restrict__ b_cc,
                           float* __restrict__ kcc) {
    const int id = blockIdx.x * 256 + threadIdx.x;   // 0 .. 8*4096-1
    if (id >= NB * NC * NC) return;
    const int b = id >> 12;
    const int r = id & 4095;
    const float* mg = meanG + b * NC;
    const float* wr = w_cc + (size_t)r * NC;
    float s = b_cc[r];
    #pragma unroll 8
    for (int g = 0; g < NC; ++g) s += mg[g] * wr[g];
    kcc[id] = s;
}

// ---------------- kernel 3: guided-kernel map via implicit-GEMM WMMA ----------------
// block = 128 threads (4 waves). blockIdx -> (b, y, ocg). Each wave: 16 pixels x 16 oc,
// K = 576 (tap-major) in 18 steps of 32 (f16 WMMA, f32 accum). All A/B fragment reads
// are contiguous 16B-aligned runs -> ds_load_b128.
__global__ __launch_bounds__(128) void conv_kgen_kernel(const float* __restrict__ guidance,
                                                        const float* __restrict__ w_kgl,
                                                        const float* __restrict__ b_kgl,
                                                        float* __restrict__ kmap) {
    __shared__ _Float16 wgt[16 * KCONV];        // [ocl][k], k = tap*64+ic   (18 KB)
    __shared__ _Float16 gt[3 * 66 * 64];        // [row r][col][ic]          (25 KB)

    const int tid  = threadIdx.x;
    const int lane = tid & 31;
    const int wv   = tid >> 5;                  // wave 0..3 -> x sub-tile
    const int blk  = blockIdx.x;                // b*322*4 + y*4 + ocg
    const int ocg  = blk & 3;
    const int y    = (blk >> 2) % HK;
    const int b    = blk / (HK * 4);

    // stage this group's weights, converting OIHW (ic*9 + tap) -> tap-major (tap*64 + ic)
    for (int i = tid; i < 16 * KCONV; i += 128) {
        const int ocl = i / KCONV;
        const int k   = i - ocl * KCONV;
        const int tap = k >> 6;
        const int ic  = k & 63;
        wgt[i] = (_Float16)w_kgl[(size_t)(ocg * 16 + ocl) * KCONV + ic * 9 + tap];
    }

    const int m    = lane & 15;   // A row (pixel) / B col (oc)
    const int hlf  = lane >> 4;
    const int w16  = wv * 16;
    const int cb6  = (w16 + m) << 6;     // (pixel col base) * 64, loop-invariant
    const int hlf8 = hlf * 8;

    for (int cc = 0; cc < 6; ++cc) {            // 6 chunks of 64 pixels cover x in [0,322)
        const int x0c = cc * 64;
        __syncthreads();
        // stage guidance rows y-2..y, cols x0c-2 .. x0c+63 (zero padded), ic innermost
        for (int i = tid; i < 3 * 66 * 64; i += 128) {
            const int ic  = i & 63;
            const int t   = i >> 6;             // 0..197
            const int col = t % 66;
            const int r   = t / 66;
            const int gy = y - 2 + r;
            const int gx = x0c - 2 + col;
            float v = 0.f;
            if ((unsigned)gy < (unsigned)HH && (unsigned)gx < (unsigned)WW)
                v = guidance[(((size_t)b * NC + ic) * HH + gy) * WW + gx];
            gt[i] = (_Float16)v;
        }
        __syncthreads();

        v8f acc = {};
        #pragma unroll
        for (int ks = 0; ks < 18; ++ks) {
            const int k0  = ks * 32;
            const int tap = k0 >> 6;            // const; runs never cross a tap block
            const int ky  = tap / 3;
            const int kx  = tap - ky * 3;
            const int ic0 = (k0 & 63) + hlf * 16;
            const int abase = ((ky * 66 + kx) << 6) + cb6 + ic0;
            // A: 16 contiguous halves as two 8-half runs (K = k0 + h*16 ..)
            const v16h afrag = cat8(*(const h8*)&gt[abase],
                                    *(const h8*)&gt[abase + 8]);
            // B: 16 contiguous halves (K = k0 + h*16 ..)
            const v16h bfrag = *(const v16h*)&wgt[m * KCONV + k0 + hlf * 16];
            acc = __builtin_amdgcn_wmma_f32_16x16x32_f16(
                false, afrag, false, bfrag, (short)0, acc, false, false);
        }

        // C/D layout: lane -> oc = ocg*16 + (lane&15); VGPR r -> pixel m = r + 8*hlf
        const int oc = ocg * 16 + m;
        const float bias = b_kgl[oc];
        const int x0 = x0c + w16;
        #pragma unroll
        for (int r = 0; r < 8; ++r) {
            const int px = x0 + r + 8 * hlf;
            if (px < WK)
                kmap[(((size_t)b * NC + oc) * HK + y) * WK + px] = acc[r] + bias;
        }
    }
}

// ---------------- kernel 4: x1 = boxsum3x3( pad(data) * kmap ) ----------------
__global__ void prodsum_kernel(const float* __restrict__ data,
                               const float* __restrict__ kmap,
                               float* __restrict__ x1) {
    const int id = blockIdx.x * 256 + threadIdx.x;
    if (id >= NTOT) return;
    const int x  = id % WW;
    int t = id / WW;
    const int y  = t % HH;
    t /= HH;
    const int c  = t % NC;
    const int b  = t / NC;
    const size_t kbase = ((size_t)b * NC + c) * HK * WK;
    const size_t dbase = ((size_t)b * NC + c) * HH * WW;
    __builtin_prefetch(&kmap[kbase + (size_t)(y + 3) * WK + x], 0, 1);
    float s = 0.f;
    #pragma unroll
    for (int ky = 0; ky < 3; ++ky) {
        #pragma unroll
        for (int kx = 0; kx < 3; ++kx) {
            const int yy = y + ky, xx = x + kx;                // kmap coords, in-range
            const float km = kmap[kbase + (size_t)yy * WK + xx];
            const int dy = yy - 1, dx = xx - 1;                // padded data coords
            float dv = 0.f;
            if ((unsigned)dy < (unsigned)HH && (unsigned)dx < (unsigned)WW)
                dv = data[dbase + (size_t)dy * WW + dx];
            s += km * dv;
        }
    }
    x1[id] = s;
}

// ---------------- kernel 5: x2 = kcc x1 (WMMA), leaky ReLU, BN-stat accumulation ----------------
__global__ __launch_bounds__(128) void ccmix_kernel(const float* __restrict__ x1,
                                                    const float* __restrict__ kcc,
                                                    float* __restrict__ out,
                                                    float* __restrict__ bnSum,
                                                    float* __restrict__ bnSqs) {
    __shared__ _Float16 kl[NC * NC];   // [oc][c]        (8 KB)
    __shared__ _Float16 al[16 * NC];   // [pixel m][c]   (2 KB)

    const int tid  = threadIdx.x;
    const int lane = tid & 31;
    const int wv   = tid >> 5;
    const int b     = blockIdx.x / 100;
    const int chunk = blockIdx.x % 100;              // 100 chunks x 64 tiles x 16 px = 102400

    // stage per-batch kcc as f16
    {
        const float* kb = kcc + (size_t)b * NC * NC;
        for (int i = tid; i < NC * NC; i += 128) kl[i] = (_Float16)kb[i];
    }

    const int nn  = lane & 15;
    const int hlf = lane >> 4;
    const int oc  = wv * 16 + nn;                    // this lane's output channel
    float s = 0.f, s2 = 0.f;

    for (int it = 0; it < 64; ++it) {
        const int p0 = (chunk * 64 + it) * 16;
        __syncthreads();
        for (int i = tid; i < 16 * NC; i += 128) {   // al[m*64+c] = x1[b][c][p0+m]
            const int mm = i >> 6;
            const int c  = i & 63;
            al[i] = (_Float16)x1[((size_t)b * NC + c) * NPIX + p0 + mm];
        }
        __syncthreads();

        v8f acc = {};
        #pragma unroll
        for (int ks = 0; ks < 2; ++ks) {
            const int k0 = ks * 32;
            const int abase = nn * NC + k0 + hlf * 16;
            const v16h afrag = cat8(*(const h8*)&al[abase],
                                    *(const h8*)&al[abase + 8]);
            const v16h bfrag = *(const v16h*)&kl[oc * NC + k0 + hlf * 16];
            acc = __builtin_amdgcn_wmma_f32_16x16x32_f16(
                false, afrag, false, bfrag, (short)0, acc, false, false);
        }

        const size_t obase = ((size_t)b * NC + oc) * NPIX;
        #pragma unroll
        for (int r = 0; r < 8; ++r) {
            const int px = p0 + r + 8 * hlf;
            float v = acc[r];
            v = (v > 0.f) ? v : 0.2f * v;            // leaky ReLU
            out[obase + px] = v;
            s  += v;
            s2 += v * v;
        }
    }
    atomicAdd(&bnSum[oc], s);
    atomicAdd(&bnSqs[oc], s2);
}

// ---------------- kernel 6: batchnorm finalize (in-place on out) ----------------
__global__ void bn_kernel(float* __restrict__ out,
                          const float* __restrict__ bnSum,
                          const float* __restrict__ bnSqs,
                          const float* __restrict__ gamma,
                          const float* __restrict__ beta) {
    const int id = blockIdx.x * 256 + threadIdx.x;
    if (id >= NTOT) return;
    const int c = (id / NPIX) % NC;
    const float n   = (float)(NB * NPIX);
    const float mu  = bnSum[c] / n;
    const float var = bnSqs[c] / n - mu * mu;
    const float inv = rsqrtf(var + 1e-5f);
    const float v = out[id];
    out[id] = (v - mu) * inv * gamma[c] + beta[c];
}

// ---------------- launcher ----------------
extern "C" void kernel_launch(void* const* d_in, const int* in_sizes, int n_in,
                              void* d_out, int out_size, void* d_ws, size_t ws_size,
                              hipStream_t stream) {
    const float* data     = (const float*)d_in[0];
    const float* guidance = (const float*)d_in[1];
    const float* w_kgl    = (const float*)d_in[2];
    const float* b_kgl    = (const float*)d_in[3];
    const float* w_cc     = (const float*)d_in[4];
    const float* b_cc     = (const float*)d_in[5];
    const float* gamma    = (const float*)d_in[6];
    const float* beta     = (const float*)d_in[7];
    float* out = (float*)d_out;

    float* ws    = (float*)d_ws;
    float* meanG = ws;                           // 512
    float* kcc   = ws + 512;                     // 32768
    float* bnSum = ws + 33280;                   // 64
    float* bnSqs = ws + 33344;                   // 64
    float* kmap  = ws + 33536;                   // 8*64*322*322 = 53,086,208
    float* x1    = ws + 33536 + (size_t)NB * NC * HK * WK;  // 52,428,800

    zero_stats_kernel<<<1, 128, 0, stream>>>(bnSum);
    meang_kernel<<<NB * NC, 256, 0, stream>>>(guidance, meanG);
    kcc_kernel<<<(NB * NC * NC + 255) / 256, 256, 0, stream>>>(meanG, w_cc, b_cc, kcc);
    conv_kgen_kernel<<<NB * HK * 4, 128, 0, stream>>>(guidance, w_kgl, b_kgl, kmap);
    prodsum_kernel<<<(NTOT + 255) / 256, 256, 0, stream>>>(data, kmap, x1);
    ccmix_kernel<<<NB * 100, 128, 0, stream>>>(x1, kcc, out, bnSum, bnSqs);
    bn_kernel<<<(NTOT + 255) / 256, 256, 0, stream>>>(out, bnSum, bnSqs, gamma, beta);
}
